// TransformerV2_53060025975283
// MI455X (gfx1250) — matrix operand order repaired and checked by reading, not compiled, not run on hardware
//
#include <hip/hip_runtime.h>
#include <math.h>

// ---------------------------------------------------------------------------
// Problem constants (B=1 fixed by the reference)
// ---------------------------------------------------------------------------
#define HW      96
#define LQ      9216          // 96*96 spatial positions
#define INC     64            // feature channels
#define QC      32            // q/k channels
#define K9      288           // QC*9 unfold depth  (== 9 * 32 WMMA k-steps)
#define TOPK    5
#define KSPLIT  8             // key-dimension split for occupancy
#define KEYS_PER_SPLIT (LQ / KSPLIT)   // 1152
#define KITERS  (KEYS_PER_SPLIT / 16)  // 72 key tiles of 16 per split

typedef __attribute__((ext_vector_type(16))) _Float16 v16h;
typedef __attribute__((ext_vector_type(8)))  _Float16 v8h;
typedef __attribute__((ext_vector_type(8)))  float    v8f;

// Async b64 chunk type: the builtin's probe-revealed signature wants
// 'int __attribute__((vector_size(8)))' pointers (global AS1 / LDS AS3).
typedef int v2i_b64 __attribute__((vector_size(8)));
#define GPTR_B64(p) ((__attribute__((address_space(1))) v2i_b64*)(p))
#define LPTR_B64(p) ((__attribute__((address_space(3))) v2i_b64*)(p))

// CDNA5 async global->LDS path (ASYNCcnt-tracked), guarded so a toolchain
// without the builtins falls back to the proven synchronous staging.
#define USE_ASYNC (__has_builtin(__builtin_amdgcn_global_load_async_to_lds_b64))

#if __has_builtin(__builtin_amdgcn_s_wait_asynccnt)
#define WAIT_ASYNC(n) __builtin_amdgcn_s_wait_asynccnt(n)
#else
#define WAIT_ASYNC(n) asm volatile("s_wait_asynccnt %0" :: "i"(n) : "memory")
#endif

// ---------------------------------------------------------------------------
// 3x3 SAME conv, stride 1, in_c = 64, optional ReLU.  ~2 GFLOP total over the
// four calls -> plain fp32 VALU is fine; x (2.4 MB) lives in L2.
// ---------------------------------------------------------------------------
__global__ __launch_bounds__(256) void conv3x3_relu(
    const float* __restrict__ x, const float* __restrict__ Wt,
    const float* __restrict__ bias, float* __restrict__ y,
    int out_c, int do_relu)
{
  int t = blockIdx.x * 256 + threadIdx.x;
  if (t >= out_c * LQ) return;
  int o = t / LQ, p = t - o * LQ;
  int h = p / HW, w = p - h * HW;
  float acc = bias[o];
  const float* wr = Wt + (size_t)o * INC * 9;
  for (int c = 0; c < INC; ++c) {
    const float* xc = x + (size_t)c * LQ;
#pragma unroll
    for (int i = 0; i < 3; ++i) {
      int hh = h + i - 1;
      if ((unsigned)hh >= (unsigned)HW) continue;
#pragma unroll
      for (int j = 0; j < 3; ++j) {
        int ww = w + j - 1;
        if ((unsigned)ww >= (unsigned)HW) continue;
        acc = fmaf(wr[c * 9 + i * 3 + j], xc[hh * HW + ww], acc);
      }
    }
  }
  if (do_relu) acc = fmaxf(acc, 0.0f);
  y[t] = acc;
}

// ---------------------------------------------------------------------------
// unfold3 + L2-normalize along the 288-dim, emit f16 rows [l][288].
// Row-major per position: exactly the contiguous layout the WMMA fragment
// loads below want (16B-aligned 32-element k-step chunks).
// ---------------------------------------------------------------------------
__global__ __launch_bounds__(256) void unfold_l2norm_f16(
    const float* __restrict__ s, _Float16* __restrict__ dst)
{
  int l = blockIdx.x * 256 + threadIdx.x;
  if (l >= LQ) return;
  int h = l / HW, w = l - h * HW;
  float ss = 0.0f;
  for (int c = 0; c < QC; ++c) {
    const float* sc = s + (size_t)c * LQ;
#pragma unroll
    for (int i = 0; i < 3; ++i) {
      int hh = h + i - 1;
      bool hv = (unsigned)hh < (unsigned)HW;
#pragma unroll
      for (int j = 0; j < 3; ++j) {
        int ww = w + j - 1;
        float v = (hv && (unsigned)ww < (unsigned)HW) ? sc[hh * HW + ww] : 0.0f;
        ss = fmaf(v, v, ss);
      }
    }
  }
  float inv = 1.0f / fmaxf(sqrtf(ss), 1e-12f);
  _Float16* dr = dst + (size_t)l * K9;
  int idx = 0;
  for (int c = 0; c < QC; ++c) {
    const float* sc = s + (size_t)c * LQ;
#pragma unroll
    for (int i = 0; i < 3; ++i) {
      int hh = h + i - 1;
      bool hv = (unsigned)hh < (unsigned)HW;
#pragma unroll
      for (int j = 0; j < 3; ++j) {
        int ww = w + j - 1;
        float v = (hv && (unsigned)ww < (unsigned)HW) ? sc[hh * HW + ww] : 0.0f;
        dr[idx++] = (_Float16)(v * inv);
      }
    }
  }
}

// ---------------------------------------------------------------------------
// Fused correlation GEMM (f16 WMMA, f32 acc) + streaming top-5.
// Never materializes the 9216x9216 R matrix (340 MB > L2).
//
// Block = 128 threads = 4 waves; wave owns 32 queries (two 16-wide B tiles,
// register resident: 2 tiles * 9 k-steps * 8 VGPRs).  16-key tiles are
// double-buffered in LDS and staged with ASYNC global->LDS loads (9 b64
// chunks per thread per tile -> uniform per-wave ASYNCcnt, so the wait
// immediate is exact).  grid = (72 query blocks, 8 key splits).
// ---------------------------------------------------------------------------
__global__ __launch_bounds__(128) void corr_topk_wmma(
    const _Float16* __restrict__ Qn, const _Float16* __restrict__ Kn,
    float* __restrict__ pv, int* __restrict__ pi)
{
  __shared__ __align__(16) _Float16 Kbuf[2][16 * K9];   // 2 x 9 KB key tiles
  __shared__ float Ssc[4][16 * 32];                     // 8 KB per-wave scores

  const int tid  = threadIdx.x;
  const int wave = tid >> 5;
  const int lane = tid & 31;
  const int nIdx = lane & 15;
  const int hiH  = lane >> 4;            // 0 for lanes 0-15, 1 for lanes 16-31

  const int qbase = blockIdx.x * 128 + wave * 32;
  const int split = blockIdx.y;
  const _Float16* KnS = Kn + (size_t)split * KEYS_PER_SPLIT * K9;
  const int keyBase0 = split * KEYS_PER_SPLIT;

  // B fragments: B is 32x16 (KxN) per k-step; lanes 0-15 hold K0..15 of column
  // N=lane, lanes 16-31 hold K16..31 -> one 32B contiguous chunk per lane.
  v16h Bf[2][9];
#pragma unroll
  for (int qt = 0; qt < 2; ++qt) {
    const _Float16* qr = Qn + (size_t)(qbase + qt * 16 + nIdx) * K9;
#pragma unroll
    for (int kk = 0; kk < 9; ++kk) {
      const v8h* p = (const v8h*)(qr + kk * 32 + hiH * 16);
      v8h lo = p[0];
      v8h hi = p[1];
      Bf[qt][kk] = __builtin_shufflevector(lo, hi,
          0, 1, 2, 3, 4, 5, 6, 7, 8, 9, 10, 11, 12, 13, 14, 15);
    }
  }

  float tv[TOPK];
  int   ti[TOPK];
#pragma unroll
  for (int i = 0; i < TOPK; ++i) { tv[i] = -3.0e30f; ti[i] = 0; }

#if USE_ASYNC
  // Prologue: stage tile 0 into Kbuf[0] (1152 x 8B = 9 chunks per thread).
  {
    char* g0 = (char*)KnS;          // builtin takes non-const AS1 pointer
    char* l0 = (char*)&Kbuf[0][0];
#pragma unroll
    for (int r = 0; r < 9; ++r) {
      int j = tid + r * 128;
      __builtin_amdgcn_global_load_async_to_lds_b64(
          GPTR_B64(g0 + j * 8), LPTR_B64(l0 + j * 8), 0, 0);
    }
  }
#endif

  for (int kt = 0; kt < KITERS; ++kt) {
#if USE_ASYNC
    const int cur = kt & 1;
    // Issue next tile into the other buffer (its last readers all passed the
    // mid-iteration barrier of kt-1), then wait for the current tile's 9
    // oldest async loads and publish across waves.
    if (kt + 1 < KITERS) {
      char* gn = (char*)(KnS + (size_t)(kt + 1) * 16 * K9);
      char* ln = (char*)&Kbuf[cur ^ 1][0];
#pragma unroll
      for (int r = 0; r < 9; ++r) {
        int j = tid + r * 128;
        __builtin_amdgcn_global_load_async_to_lds_b64(
            GPTR_B64(gn + j * 8), LPTR_B64(ln + j * 8), 0, 0);
      }
      WAIT_ASYNC(9);
    } else {
      WAIT_ASYNC(0);
    }
    __syncthreads();
    const _Float16* ktile = &Kbuf[cur][0];
#else
    __syncthreads();   // previous iteration done consuming Kbuf[0] / Ssc
    {
      const float4* src = (const float4*)(KnS + (size_t)kt * 16 * K9);
      float4* dst = (float4*)&Kbuf[0][0];
      for (int i = tid; i < (16 * K9) / 8; i += 128) dst[i] = src[i];
    }
    if (kt + 1 < KITERS) {  // pull next tile toward near caches
      const char* np = (const char*)(KnS + (size_t)(kt + 1) * 16 * K9);
      __builtin_prefetch(np + tid * 72, 0, 3);
    }
    __syncthreads();
    const _Float16* ktile = &Kbuf[0][0];
#endif

    // A fragment: 16x32 (MxK) f16; lane row M = lane%16;
    // lanes 0-15: K0..15, lanes 16-31: K16..31 (matches B).
    v8f acc0 = {};
    v8f acc1 = {};
    const _Float16* ar = ktile + (size_t)nIdx * K9;
#pragma unroll
    for (int kk = 0; kk < 9; ++kk) {
      v8h alo = *(const v8h*)(ar + kk * 32 + hiH * 16);
      v8h ahi = *(const v8h*)(ar + kk * 32 + hiH * 16 + 8);
      v16h a = __builtin_shufflevector(alo, ahi,
          0, 1, 2, 3, 4, 5, 6, 7, 8, 9, 10, 11, 12, 13, 14, 15);
      acc0 = __builtin_amdgcn_wmma_f32_16x16x32_f16(
          false, a, false, Bf[0][kk], (short)0, acc0, false, false);
      acc1 = __builtin_amdgcn_wmma_f32_16x16x32_f16(
          false, a, false, Bf[1][kk], (short)0, acc1, false, false);
    }

    // Spill 16(keys) x 32(queries) scores to per-wave LDS scratch.
    // D layout: VGPR i -> row M=i (lanes 0-15) / M=i+8 (lanes 16-31), col=N.
    float* Sw = Ssc[wave];
#pragma unroll
    for (int i = 0; i < 8; ++i) {
      int m = i + 8 * hiH;
      Sw[m * 32 + nIdx]      = acc0[i];
      Sw[m * 32 + 16 + nIdx] = acc1[i];
    }
    __syncthreads();

    // Each lane owns one query column; streaming sorted top-5 insertion.
    // Strict '>' + ascending key order == jax top_k tie rule (lower idx wins).
    const int keyBase = keyBase0 + kt * 16;
#pragma unroll
    for (int m = 0; m < 16; ++m) {
      float v = Sw[m * 32 + lane];
      if (v > tv[TOPK - 1]) {
        tv[TOPK - 1] = v;
        ti[TOPK - 1] = keyBase + m;
#pragma unroll
        for (int s = TOPK - 1; s > 0; --s) {
          if (tv[s] > tv[s - 1]) {
            float tf = tv[s]; tv[s] = tv[s - 1]; tv[s - 1] = tf;
            int   tn = ti[s]; ti[s] = ti[s - 1]; ti[s - 1] = tn;
          }
        }
      }
    }
  }

  const int q = qbase + lane;
#pragma unroll
  for (int i = 0; i < TOPK; ++i) {
    pv[(size_t)(split * TOPK + i) * LQ + q] = tv[i];
    pi[(size_t)(split * TOPK + i) * LQ + q] = ti[i];
  }
}

// ---------------------------------------------------------------------------
// Merge the KSPLIT partial top-5 lists into the global top-5 per query.
// Candidates are scanned in (split asc, rank asc) order, which is globally
// index-ascending among equal values -> first-hit-wins matches top_k ties.
// ---------------------------------------------------------------------------
__global__ __launch_bounds__(256) void topk_merge(
    const float* __restrict__ pv, const int* __restrict__ pi,
    float* __restrict__ tv, int* __restrict__ ti)
{
  int q = blockIdx.x * 256 + threadIdx.x;
  if (q >= LQ) return;
  float cv[KSPLIT * TOPK];
  int   ci[KSPLIT * TOPK];
#pragma unroll
  for (int n = 0; n < KSPLIT * TOPK; ++n) {
    cv[n] = pv[(size_t)n * LQ + q];
    ci[n] = pi[(size_t)n * LQ + q];
  }
#pragma unroll
  for (int o = 0; o < TOPK; ++o) {
    float best = -3.0e30f;
    int bidx = 0;
#pragma unroll
    for (int n = 0; n < KSPLIT * TOPK; ++n)
      if (cv[n] > best) { best = cv[n]; bidx = ci[n]; }
    tv[(size_t)o * LQ + q] = best;
    ti[(size_t)o * LQ + q] = bidx;
#pragma unroll
    for (int n = 0; n < KSPLIT * TOPK; ++n)     // indices are globally unique
      if (cv[n] == best && ci[n] == bidx) cv[n] = -3.0e30f;
  }
}

// ---------------------------------------------------------------------------
// acc = sum_{t=1..4} S_t * fold3(gather(x_unfold, idx_t))/9 ; y = feat+x+acc/4
// Gathers hit x (2.4 MB, L2-resident).
// ---------------------------------------------------------------------------
__global__ __launch_bounds__(256) void gather_fold_accum(
    const float* __restrict__ x, const float* __restrict__ feat,
    const float* __restrict__ tv, const int* __restrict__ ti,
    float* __restrict__ y)
{
  int t = blockIdx.x * 256 + threadIdx.x;
  if (t >= INC * LQ) return;
  int c = t / LQ, p = t - c * LQ;
  int h = p / HW, w = p - h * HW;
  const float* xc = x + (size_t)c * LQ;
  float acc = 0.0f;
#pragma unroll
  for (int tk = 1; tk < TOPK; ++tk) {
    float s = tv[(size_t)tk * LQ + p];
    const int* tix = ti + (size_t)tk * LQ;
    float fsum = 0.0f;
#pragma unroll
    for (int i = 0; i < 3; ++i) {
      int hp = h + 1 - i;
      if ((unsigned)hp >= (unsigned)HW) continue;
#pragma unroll
      for (int j = 0; j < 3; ++j) {
        int wp = w + 1 - j;
        if ((unsigned)wp >= (unsigned)HW) continue;
        int m = tix[hp * HW + wp];
        int mr = m / HW;
        int r  = mr + i - 1;
        int sc = (m - mr * HW) + j - 1;
        if ((unsigned)r < (unsigned)HW && (unsigned)sc < (unsigned)HW)
          fsum += xc[r * HW + sc];
      }
    }
    acc = fmaf(s, fsum * (1.0f / 9.0f), acc);
  }
  y[t] = feat[t] + xc[p] + acc * 0.25f;
}

// ---------------------------------------------------------------------------
extern "C" void kernel_launch(void* const* d_in, const int* in_sizes, int n_in,
                              void* d_out, int out_size, void* d_ws, size_t ws_size,
                              hipStream_t stream)
{
  const float* x  = (const float*)d_in[0];
  const float* W1 = (const float*)d_in[1];
  const float* b1 = (const float*)d_in[2];
  const float* W2 = (const float*)d_in[3];
  const float* b2 = (const float*)d_in[4];
  const float* Wq = (const float*)d_in[5];
  const float* bq = (const float*)d_in[6];
  const float* Wk = (const float*)d_in[7];
  const float* bk = (const float*)d_in[8];
  float* y = (float*)d_out;

  // Workspace carve (all chunks 16B aligned); total ~21 MB.
  char* ws = (char*)d_ws;
  float*    t1   = (float*)ws;     ws += (size_t)INC * LQ * 4;
  float*    feat = (float*)ws;     ws += (size_t)INC * LQ * 4;
  float*    qf   = (float*)ws;     ws += (size_t)QC * LQ * 4;
  float*    kf   = (float*)ws;     ws += (size_t)QC * LQ * 4;
  _Float16* Qn   = (_Float16*)ws;  ws += (size_t)LQ * K9 * 2;
  _Float16* Kn   = (_Float16*)ws;  ws += (size_t)LQ * K9 * 2;
  float*    pv   = (float*)ws;     ws += (size_t)KSPLIT * TOPK * LQ * 4;
  int*      pi   = (int*)ws;       ws += (size_t)KSPLIT * TOPK * LQ * 4;
  float*    tv   = (float*)ws;     ws += (size_t)TOPK * LQ * 4;
  int*      ti   = (int*)ws;       ws += (size_t)TOPK * LQ * 4;

  conv3x3_relu<<<(INC * LQ + 255) / 256, 256, 0, stream>>>(x,  W1, b1, t1,   INC, 1);
  conv3x3_relu<<<(INC * LQ + 255) / 256, 256, 0, stream>>>(t1, W2, b2, feat, INC, 1);
  conv3x3_relu<<<(QC  * LQ + 255) / 256, 256, 0, stream>>>(x,  Wq, bq, qf,   QC, 0);
  conv3x3_relu<<<(QC  * LQ + 255) / 256, 256, 0, stream>>>(x,  Wk, bk, kf,   QC, 0);

  unfold_l2norm_f16<<<(LQ + 255) / 256, 256, 0, stream>>>(qf, Qn);
  unfold_l2norm_f16<<<(LQ + 255) / 256, 256, 0, stream>>>(kf, Kn);

  corr_topk_wmma<<<dim3(LQ / 128, KSPLIT), 128, 0, stream>>>(Qn, Kn, pv, pi);
  topk_merge<<<(LQ + 255) / 256, 256, 0, stream>>>(pv, pi, tv, ti);

  gather_fold_accum<<<(INC * LQ + 255) / 256, 256, 0, stream>>>(x, feat, tv, ti, y);
}
